// ESNForce_5514738008251
// MI455X (gfx1250) — matrix operand order repaired and checked by reading, not compiled, not run on hardware
//
#include <hip/hip_runtime.h>
#include <hip/hip_bf16.h>

// ESN rollout for MI455X (gfx1250).
// W_res (64MB fp32) -> bf16 WMMA-swizzled tiles (32MB, resident in the 192MB
// L2). Per step: kernel1 = state update, one wave per 16 rows (256 blocks x
// 1 wave -> one strip per WGP), 128 k-tiles of 32 each doing one
// v_wmma_f32_16x16x32_bf16 (A = pre-swizzled W_res tile from global/L2,
// B = state vector broadcast from LDS), fused input/feedback/bias/tanh/leaky
// tail; kernel2 = readout y = W_out @ s on a full WGP (1024 threads).
// The 4096-step serial dependence is honored via stream ordering.

#define ESN_H 4096
#define ESN_I 64
#define ESN_O 16
#define ESN_T 4096
#define ESN_LEAK 0.9f

typedef __attribute__((ext_vector_type(16))) __bf16        v16bf;
typedef __attribute__((ext_vector_type(8)))  float         v8f;
typedef __attribute__((ext_vector_type(8)))  unsigned int  v8u;

__device__ __forceinline__ unsigned short esn_f2bf(float f) {
    unsigned int u = __builtin_bit_cast(unsigned int, f);
    u += 0x7FFFu + ((u >> 16) & 1u);          // round-to-nearest-even
    return (unsigned short)(u >> 16);
}

// ---------------------------------------------------------------------------
// Prep: convert W_res (fp32 row-major [H,H]) into bf16 WMMA-A-swizzled tiles.
// Tile (mt,kt) = rows mt*16..+15, cols kt*32..+31. Storage: tile-major,
// then lane (0..31), then 16 halves = the lane's A VGPRs 0..7 (2 halves each).
// A layout used (16-bit 16x32, contiguous k): lane L -> M = L&15, hi = L>>4;
//   VGPR v : K = 16*hi + 2v + p (same k order as the B slice from LDS)
// ---------------------------------------------------------------------------
__global__ __launch_bounds__(256) void esn_prep_swizzle(
    const float* __restrict__ Wres, unsigned short* __restrict__ dst) {
    size_t tid = (size_t)blockIdx.x * 256 + threadIdx.x;    // H*H threads
    unsigned tile   = (unsigned)(tid >> 9);                  // 512 halves/tile
    unsigned within = (unsigned)(tid & 511);
    unsigned lane = within >> 4, hidx = within & 15;
    unsigned v = hidx >> 1, p = hidx & 1;
    unsigned hi = lane >> 4, m = lane & 15;
    unsigned k = (hi << 4) + (v << 1) + p;
    unsigned mt = tile >> 7, kt = tile & 127;
    size_t row = (size_t)mt * 16 + m;
    size_t col = (size_t)kt * 32 + k;
    dst[tid] = esn_f2bf(Wres[row * ESN_H + col]);
}

__global__ __launch_bounds__(256) void esn_init(
    float* __restrict__ s_f32, unsigned short* __restrict__ s_bf2,
    float* __restrict__ y) {
    int tid = blockIdx.x * 256 + threadIdx.x;
    if (tid < ESN_H) {
        s_f32[tid] = 0.0f;
        s_bf2[tid] = 0;            // buffer 0
        s_bf2[ESN_H + tid] = 0;    // buffer 1
    }
    if (tid < ESN_O) y[tid] = 0.0f;
}

// ---------------------------------------------------------------------------
// Step kernel 1: s_new = (1-a)*s + a*tanh(Wres@s + Win@x_t + b + Wfb@y)
// 256 blocks x 32 threads: one wave per WGP owns 16 rows, loops 128 k-tiles
// of 32 with one bf16 WMMA each (f32 accumulate).
// ---------------------------------------------------------------------------
__global__ __launch_bounds__(32) void esn_step_state(
    const unsigned short* __restrict__ wres_sw,  // swizzled bf16 [H*H]
    const unsigned short* __restrict__ s_bf_in,  // [H] state, bf16 (read)
    unsigned short*       __restrict__ s_bf_out, // [H] state, bf16 (write)
    float*                             s_f32,    // [H] state fp32 (r/w, row-exclusive)
    const float* __restrict__ x_t,               // [I]
    const float* __restrict__ Win,               // [H,I]
    const float* __restrict__ bias,              // [H]
    const float* __restrict__ Wfb,               // [H,O]
    const float* __restrict__ y_prev,            // [O]
    float*       __restrict__ out_state)         // outputs + t*H
{
    __shared__ unsigned short s_lds[ESN_H];      // 8KB: state as bf16, broadcast B
    __shared__ float raw[16];                    // raw Wres@s rows for this wave

    const int lane = threadIdx.x;                // 0..31, single wave

    // stage bf16 state into LDS (whole vector; every k-tile reads from it)
    {
        const uint4* src = (const uint4*)s_bf_in;
        uint4*       dst = (uint4*)s_lds;
        #pragma unroll
        for (int i = 0; i < 16; ++i) dst[lane + i * 32] = src[lane + i * 32];
    }
    __syncthreads();

    const int mt = blockIdx.x;                   // row strip: rows mt*16..+15
    const v8u* __restrict__ atiles =
        (const v8u*)wres_sw + (size_t)mt * 128 * 32;   // 32 v8u (=1KB) per tile

    v8f acc = {};
    const int hoff = (lane >> 4) << 4;           // 0 or 16 halves into k-tile

    for (int kt = 0; kt < 128; ++kt) {
        const v8u* ap = &atiles[(size_t)kt * 32 + lane];
        // Prefetch 4 tiles (4KB) ahead at default WGP scope: per ISA 10.5 only
        // scope 0 pulls into the WGP-near caches (SE/DEV prefetch stops at GL2,
        // where W_res already resides). Emit via inline asm since the builtin
        // only reaches SE/DEV scopes.
        asm volatile("global_prefetch_b8 %0, off offset:4096" :: "v"(ap));
        // A: this lane's 16 bf16 values of tile (mt,kt), pre-swizzled
        v8u araw = *ap;
        // B: broadcast state slice s[kt*32 + hoff .. +15] from LDS
        v8u braw = *(const v8u*)&s_lds[kt * 32 + hoff];
        acc = __builtin_amdgcn_wmma_f32_16x16x32_bf16(
            false, __builtin_bit_cast(v16bf, araw),
            false, __builtin_bit_cast(v16bf, braw),
            (short)0, acc, false, false);
    }

    // D layout: lane 0 VGPR v -> row v ; lane 16 VGPR v -> row v+8 (any N copy)
    if (lane == 0) {
        #pragma unroll
        for (int v = 0; v < 8; ++v) raw[v] = acc[v];
    } else if (lane == 16) {
        #pragma unroll
        for (int v = 0; v < 8; ++v) raw[8 + v] = acc[v];
    }
    __syncthreads();

    // fused tail: lanes 0..15 each finish one row
    if (lane < 16) {
        const int row = mt * 16 + lane;
        float pre = raw[lane] + bias[row];
        const float* __restrict__ wi = Win + (size_t)row * ESN_I;
        #pragma unroll 8
        for (int j = 0; j < ESN_I; ++j) pre += wi[j] * x_t[j];
        const float* __restrict__ wf = Wfb + (size_t)row * ESN_O;
        #pragma unroll
        for (int j = 0; j < ESN_O; ++j) pre += wf[j] * y_prev[j];
        float sn = (1.0f - ESN_LEAK) * s_f32[row] + ESN_LEAK * tanhf(pre);
        s_f32[row]     = sn;
        s_bf_out[row]  = esn_f2bf(sn);
        out_state[row] = sn;
    }
}

// ---------------------------------------------------------------------------
// Step kernel 2: y = W_out @ s_new  (16 x 4096 matvec), one full WGP.
// 1024 threads: 64 lanes cooperate per output row, tree-reduced in LDS.
// ---------------------------------------------------------------------------
__global__ __launch_bounds__(1024) void esn_step_readout(
    const float* __restrict__ s_f32,
    const float* __restrict__ Wout,      // [O,H]
    float* __restrict__ y,               // [O]
    float* __restrict__ res_t)           // results + t*O
{
    __shared__ float part[1024];
    const int tid = threadIdx.x;
    const int o = tid >> 6;              // output row 0..15
    const int g = tid & 63;              // 64 lanes per row
    const float* __restrict__ w = Wout + (size_t)o * ESN_H;
    float p = 0.0f;
    for (int k = 0; k < ESN_H / 64; ++k) {
        int j = g + (k << 6);            // coalesced within each group
        p += w[j] * s_f32[j];
    }
    part[tid] = p;
    __syncthreads();
    #pragma unroll
    for (int stride = 32; stride >= 1; stride >>= 1) {
        if (g < stride) part[tid] += part[tid + stride];
        __syncthreads();
    }
    if (g == 0) {
        y[o] = part[tid];
        res_t[o] = part[tid];
    }
}

// ---------------------------------------------------------------------------
extern "C" void kernel_launch(void* const* d_in, const int* in_sizes, int n_in,
                              void* d_out, int out_size, void* d_ws, size_t ws_size,
                              hipStream_t stream) {
    const float* X    = (const float*)d_in[0];   // [T,I]
    const float* Win  = (const float*)d_in[1];   // [H,I]
    const float* Wres = (const float*)d_in[2];   // [H,H]
    const float* bias = (const float*)d_in[3];   // [H]
    const float* Wfb  = (const float*)d_in[4];   // [H,O]
    const float* Wout = (const float*)d_in[5];   // [O,H]

    float* results = (float*)d_out;                          // [T,O]
    float* outputs = (float*)d_out + (size_t)ESN_T * ESN_O;  // [T,H]

    // workspace layout (~32.03 MB)
    char* ws = (char*)d_ws;
    unsigned short* wres_sw = (unsigned short*)ws;                       // H*H bf16
    size_t off = (size_t)ESN_H * ESN_H * sizeof(unsigned short);
    float* s_f32 = (float*)(ws + off);            off += ESN_H * sizeof(float);
    unsigned short* s_bf2 = (unsigned short*)(ws + off);
    off += 2 * (size_t)ESN_H * sizeof(unsigned short);                   // double buffer
    float* y = (float*)(ws + off);
    (void)ws_size; (void)in_sizes; (void)n_in; (void)out_size;

    // one-time prep (re-run every call: deterministic)
    esn_prep_swizzle<<<(ESN_H * ESN_H) / 256, 256, 0, stream>>>(Wres, wres_sw);
    esn_init<<<(ESN_H + 255) / 256, 256, 0, stream>>>(s_f32, s_bf2, y);

    for (int t = 0; t < ESN_T; ++t) {
        const unsigned short* sbf_in  = s_bf2 + (size_t)(t & 1) * ESN_H;
        unsigned short*       sbf_out = s_bf2 + (size_t)((t + 1) & 1) * ESN_H;
        esn_step_state<<<ESN_H / 16, 32, 0, stream>>>(
            wres_sw, sbf_in, sbf_out, s_f32,
            X + (size_t)t * ESN_I, Win, bias, Wfb, y,
            outputs + (size_t)t * ESN_H);
        esn_step_readout<<<1, 1024, 0, stream>>>(
            s_f32, Wout, y, results + (size_t)t * ESN_O);
    }
}
